// Feature_Embeeding_1073741824206
// MI455X (gfx1250) — matrix operand order repaired and checked by reading, not compiled, not run on hardware
//
#include <hip/hip_runtime.h>
#include <hip/hip_bf16.h>
#include <math.h>

// ---------------- problem constants (from reference) ----------------
#define BB    4
#define NPTS  1024
#define KNN   64
#define RAD2  0.09f          // RADIUS^2
#define EPS_GN 1e-5f

// ---------------- WMMA types ----------------
typedef __attribute__((ext_vector_type(16))) __bf16 v16bf;
typedef __attribute__((ext_vector_type(8)))  float  v8f;

union ABt { v16bf v; uint4 q[2]; };               // 32B: one WMMA A or B operand
union PK8 { __hip_bfloat16 h[8]; uint4 q; };      // 16B: 8 packed bf16

// =====================================================================
// Ball query: one wave per query point, ordered append via ballot rank.
// Matches reference: indices m with d2 <= R^2 in increasing order, first
// KNN kept, padded with the first in-radius index.
// =====================================================================
__global__ __launch_bounds__(256) void ball_query_k(const float* __restrict__ xyz,
                                                    int* __restrict__ idx)
{
    int gid  = blockIdx.x * blockDim.x + threadIdx.x;
    int wave = gid >> 5;
    int lane = threadIdx.x & 31;
    if (wave >= BB * NPTS) return;
    int b = wave >> 10;
    int n = wave & (NPTS - 1);
    const float* base = xyz + (size_t)b * NPTS * 3;
    float qx = base[n*3+0], qy = base[n*3+1], qz = base[n*3+2];
    int* orow = idx + (size_t)wave * KNN;

    int found = 0;
    int first = n;
    for (int m0 = 0; m0 < NPTS && found < KNN; m0 += 32) {
        int m = m0 + lane;
        float dx = base[m*3+0] - qx;
        float dy = base[m*3+1] - qy;
        float dz = base[m*3+2] - qz;
        float d2 = dx*dx + dy*dy + dz*dz;
        bool in = (d2 <= RAD2);
        unsigned mask = (unsigned)__ballot(in);
        if (found == 0 && mask) first = m0 + (__ffs(mask) - 1);
        if (in) {
            int slot = found + __popc(mask & ((1u << lane) - 1u));
            if (slot < KNN) orow[slot] = m;
        }
        found += __popc(mask);
    }
    if (found > KNN) found = KNN;
    for (int k = found + lane; k < KNN; k += 32) orow[k] = first;
}

// =====================================================================
// PPF fused features -> bf16 activation buffer [col][32], col = b*K*N + k*N + n
// channels: xyz(3), d(3), a1,a2,a3, dist, gaussian  (11), padded to 32.
// =====================================================================
__device__ inline float angle3(float ux,float uy,float uz,float vx,float vy,float vz)
{
    float cx = uy*vz - uz*vy;
    float cy = uz*vx - ux*vz;
    float cz = ux*vy - uy*vx;
    float cn = sqrtf(cx*cx + cy*cy + cz*cz + 1e-12f);
    float dp = ux*vx + uy*vy + uz*vz;
    return atan2f(cn, dp);
}

__global__ __launch_bounds__(256) void fused_features_k(const float* __restrict__ xyz,
                                                        const float* __restrict__ nrm,
                                                        const int*  __restrict__ idx,
                                                        __hip_bfloat16* __restrict__ F)
{
    int t = blockIdx.x * blockDim.x + threadIdx.x;
    if (t >= BB * NPTS * KNN) return;
    int k = t & (KNN - 1);
    int n = (t >> 6) & (NPTS - 1);
    int b = t >> 16;
    int i = idx[t];

    const float* xb = xyz + (size_t)b * NPTS * 3;
    const float* nb = nrm + (size_t)b * NPTS * 3;
    float px = xb[n*3], py = xb[n*3+1], pz = xb[n*3+2];
    float gx = xb[i*3], gy = xb[i*3+1], gz = xb[i*3+2];
    float nx = nb[n*3], ny = nb[n*3+1], nz = nb[n*3+2];
    float mx = nb[i*3], my = nb[i*3+1], mz = nb[i*3+2];

    float dx = gx - px, dy = gy - py, dz = gz - pz;
    float d2 = dx*dx + dy*dy + dz*dz;
    float dist = sqrtf(d2 + 1e-12f);
    float a1 = angle3(nx,ny,nz, dx,dy,dz);
    float a2 = angle3(mx,my,mz, dx,dy,dz);
    float a3 = angle3(nx,ny,nz, mx,my,mz);
    float ew = expf(-dist*dist / RAD2);

    float f[11] = {px,py,pz, dx,dy,dz, a1,a2,a3, dist, ew};
    PK8 p[4];
    #pragma unroll
    for (int c = 0; c < 32; ++c) {
        float v = (c < 11) ? f[c] : 0.f;
        p[c >> 3].h[c & 7] = __float2bfloat16(v);
    }
    size_t col = ((size_t)b * KNN + k) * NPTS + n;   // spatial s = k*N + n
    uint4* dst = (uint4*)(F + col * 32);
    dst[0] = p[0].q; dst[1] = p[1].q; dst[2] = p[2].q; dst[3] = p[3].q;
}

// =====================================================================
// Weight f32 -> bf16 (with Cin zero-padding for layer 1)
// =====================================================================
__global__ void cvt_w_k(const float* __restrict__ src, __hip_bfloat16* __restrict__ dst,
                        int Cout, int CinSrc, int CinDst)
{
    int t = blockIdx.x * blockDim.x + threadIdx.x;
    if (t >= Cout * CinDst) return;
    int c = t % CinDst, o = t / CinDst;
    float v = (c < CinSrc) ? src[o * CinSrc + c] : 0.f;
    dst[t] = __float2bfloat16(v);
}

// =====================================================================
// WMMA GEMM: out[col][cout] = sum_cin wt[cout][cin]*act[col][cin] + bias
// act: bf16 [cols][Cin] (channel-contiguous) ; wt: bf16 [Cout][Cin]
// Block = 8 waves (2 cout-waves x 4 col-waves); wave tile = 64 cout x 64
// cols (4 M-tiles x 4 N-tiles = 16 WMMAs per K-step of 32 -> 16 b128
// loads per 16 WMMAs, ~32 FLOP/byte from cache).
// A lane map (16-bit A 16x32): lane<16 -> K[k0..k0+7]+[k0+16..k0+23],
//                              lane>=16 -> K[k0+8..]+[k0+24..].
// B lane map (32x16):          lane<16 -> K[k0..k0+7]+[k0+16..k0+23] of col N=lane,
//                              lane>=16 -> K[k0+8..]+[k0+24..] of col N=lane-16.
// =====================================================================
template<bool F32OUT>
__global__ __launch_bounds__(256) void gemm_wmma_k(const __hip_bfloat16* __restrict__ act,
                                                   const __hip_bfloat16* __restrict__ wt,
                                                   const float* __restrict__ bias,
                                                   void* __restrict__ outp,
                                                   int Cin, int Cout)
{
    int lane = threadIdx.x & 31;
    int wid  = threadIdx.x >> 5;
    int wm = wid & 1, wn = wid >> 1;
    int    c0   = blockIdx.y * 128 + wm * 64;
    size_t col0 = (size_t)blockIdx.x * 256 + (size_t)wn * 64;

    int nj     = lane & 15;
    int khalfB = (lane >> 4) << 3;   // 0 or 8
    int koffA  = (lane >> 4) << 3;   // 0 or 8

    v8f acc[4][4] = {};
    for (int k0 = 0; k0 < Cin; k0 += 32) {
        if (k0 + 32 < Cin)
            __builtin_prefetch(act + (col0 + nj) * (size_t)Cin + k0 + 32, 0, 1);
        ABt Bv[4];
        #pragma unroll
        for (int nt = 0; nt < 4; ++nt) {
            const uint4* bp = (const uint4*)(act + (col0 + (size_t)(nt*16 + nj)) * Cin
                                                 + k0 + khalfB);
            Bv[nt].q[0] = bp[0];
            Bv[nt].q[1] = bp[2];
        }
        ABt Av[4];
        #pragma unroll
        for (int mt = 0; mt < 4; ++mt) {
            const __hip_bfloat16* ap = wt + (size_t)(c0 + mt*16 + nj) * Cin + k0;
            Av[mt].q[0] = *(const uint4*)(ap + koffA);
            Av[mt].q[1] = *(const uint4*)(ap + 16 + koffA);
        }
        #pragma unroll
        for (int mt = 0; mt < 4; ++mt)
            #pragma unroll
            for (int nt = 0; nt < 4; ++nt)
                acc[mt][nt] = __builtin_amdgcn_wmma_f32_16x16x32_bf16(
                    false, Av[mt].v, false, Bv[nt].v,
                    (short)0, acc[mt][nt], false, false);
    }

    // Epilogue. D layout: VGPR r: lane<16 -> (M=r, N=lane); lane>=16 -> (M=r+8, N=lane-16)
    int chHalf = (lane >> 4) << 3;
    #pragma unroll
    for (int mt = 0; mt < 4; ++mt) {
        int chBase = c0 + mt*16 + chHalf;
        float4 b0 = *(const float4*)(bias + chBase);
        float4 b1 = *(const float4*)(bias + chBase + 4);
        float bb[8] = {b0.x,b0.y,b0.z,b0.w, b1.x,b1.y,b1.z,b1.w};
        #pragma unroll
        for (int nt = 0; nt < 4; ++nt) {
            size_t col = col0 + nt*16 + nj;
            if (F32OUT) {
                float* o = (float*)outp + col * Cout + chBase;
                float4 r0, r1;
                r0.x = acc[mt][nt][0] + bb[0]; r0.y = acc[mt][nt][1] + bb[1];
                r0.z = acc[mt][nt][2] + bb[2]; r0.w = acc[mt][nt][3] + bb[3];
                r1.x = acc[mt][nt][4] + bb[4]; r1.y = acc[mt][nt][5] + bb[5];
                r1.z = acc[mt][nt][6] + bb[6]; r1.w = acc[mt][nt][7] + bb[7];
                *(float4*)o       = r0;
                *(float4*)(o + 4) = r1;
            } else {
                PK8 p;
                #pragma unroll
                for (int r = 0; r < 8; ++r)
                    p.h[r] = __float2bfloat16(acc[mt][nt][r] + bb[r]);
                *(uint4*)((__hip_bfloat16*)outp + col * Cout + chBase) = p.q;
            }
        }
    }
}

// =====================================================================
// GroupNorm stats: sum & sumsq per (batch, group). One block per
// (colChunk, group, batch); LDS tree reduce; one f64 atomicAdd pair.
// =====================================================================
__global__ __launch_bounds__(256) void gn_stats_k(const __hip_bfloat16* __restrict__ buf,
                                                  double* __restrict__ stats,
                                                  int C, int cpg, int colsChunk,
                                                  size_t colsPerBatch)
{
    int b = blockIdx.z, g = blockIdx.y;
    size_t colBase = (size_t)b * colsPerBatch + (size_t)blockIdx.x * colsChunk;
    int cBase = g * cpg;
    int vpc = cpg >> 3;
    int totalVec = colsChunk * vpc;
    float s = 0.f, s2 = 0.f;
    for (int t = threadIdx.x; t < totalVec; t += blockDim.x) {
        int colOff = t / vpc;
        int cv = (t - colOff * vpc) << 3;
        PK8 u; u.q = *(const uint4*)(buf + (colBase + colOff) * C + cBase + cv);
        #pragma unroll
        for (int i = 0; i < 8; ++i) {
            float x = __bfloat162float(u.h[i]);
            s += x; s2 += x * x;
        }
    }
    __shared__ float sA[256], sB[256];
    int tid = threadIdx.x;
    sA[tid] = s; sB[tid] = s2;
    __syncthreads();
    for (int st = 128; st > 0; st >>= 1) {
        if (tid < st) { sA[tid] += sA[tid + st]; sB[tid] += sB[tid + st]; }
        __syncthreads();
    }
    if (tid == 0) {
        atomicAdd(&stats[((size_t)b * 8 + g) * 2 + 0], (double)sA[0]);
        atomicAdd(&stats[((size_t)b * 8 + g) * 2 + 1], (double)sB[0]);
    }
}

// =====================================================================
// GroupNorm normalize + affine + ReLU, in place, 8 channels / thread.
// =====================================================================
__global__ void gn_norm_relu_k(__hip_bfloat16* __restrict__ buf,
                               const double* __restrict__ stats,
                               const float* __restrict__ gamma,
                               const float* __restrict__ beta,
                               int C, int cpg, size_t colsPerBatch,
                               size_t totalCols, float invCount)
{
    size_t v = (size_t)blockIdx.x * blockDim.x + threadIdx.x;
    int vpc = C >> 3;
    size_t totalVec = totalCols * vpc;
    if (v >= totalVec) return;
    size_t col = v / vpc;
    int c = (int)(v - col * vpc) << 3;
    size_t b = col / colsPerBatch;
    int g = c / cpg;
    float mean = (float)stats[(b * 8 + g) * 2 + 0] * invCount;
    float var  = (float)stats[(b * 8 + g) * 2 + 1] * invCount - mean * mean;
    float inv  = rsqrtf(var + EPS_GN);
    float4 g0 = *(const float4*)(gamma + c); float4 g1 = *(const float4*)(gamma + c + 4);
    float4 e0 = *(const float4*)(beta  + c); float4 e1 = *(const float4*)(beta  + c + 4);
    float gg[8] = {g0.x,g0.y,g0.z,g0.w, g1.x,g1.y,g1.z,g1.w};
    float ee[8] = {e0.x,e0.y,e0.z,e0.w, e1.x,e1.y,e1.z,e1.w};
    PK8 u; u.q = *(uint4*)(buf + col * C + c);
    #pragma unroll
    for (int i = 0; i < 8; ++i) {
        float x = __bfloat162float(u.h[i]);
        float y = (x - mean) * inv * gg[i] + ee[i];
        u.h[i] = __float2bfloat16(y > 0.f ? y : 0.f);
    }
    *(uint4*)(buf + col * C + c) = u.q;
}

// =====================================================================
// Layer-3: normalize + ReLU fused with max over K -> [B*N][512] bf16
// =====================================================================
__global__ void gn3_max_k(const __hip_bfloat16* __restrict__ buf3,
                          const double* __restrict__ stats,
                          const float* __restrict__ gamma,
                          const float* __restrict__ beta,
                          __hip_bfloat16* __restrict__ out4, float invCount)
{
    int t = blockIdx.x * blockDim.x + threadIdx.x;
    if (t >= BB * NPTS * 64) return;           // 512/8 = 64 vectors per point
    int c = (t & 63) << 3;
    int n = (t >> 6) & (NPTS - 1);
    int b = t >> 16;
    int g = c >> 6;                            // cpg = 64
    float mean = (float)stats[((size_t)b * 8 + g) * 2 + 0] * invCount;
    float var  = (float)stats[((size_t)b * 8 + g) * 2 + 1] * invCount - mean * mean;
    float inv  = rsqrtf(var + EPS_GN);
    float4 g0 = *(const float4*)(gamma + c); float4 g1 = *(const float4*)(gamma + c + 4);
    float4 e0 = *(const float4*)(beta  + c); float4 e1 = *(const float4*)(beta  + c + 4);
    float gg[8] = {g0.x,g0.y,g0.z,g0.w, g1.x,g1.y,g1.z,g1.w};
    float ee[8] = {e0.x,e0.y,e0.z,e0.w, e1.x,e1.y,e1.z,e1.w};
    float mx[8] = {0,0,0,0,0,0,0,0};           // relu output >= 0
    for (int k = 0; k < KNN; ++k) {
        size_t col = ((size_t)b * KNN + k) * NPTS + n;
        PK8 u; u.q = *(const uint4*)(buf3 + col * 512 + c);
        #pragma unroll
        for (int i = 0; i < 8; ++i) {
            float x = __bfloat162float(u.h[i]);
            float y = (x - mean) * inv * gg[i] + ee[i];
            y = y > 0.f ? y : 0.f;
            mx[i] = fmaxf(mx[i], y);
        }
    }
    PK8 o;
    #pragma unroll
    for (int i = 0; i < 8; ++i) o.h[i] = __float2bfloat16(mx[i]);
    *(uint4*)(out4 + ((size_t)b * NPTS + n) * 512 + c) = o.q;
}

// =====================================================================
// Host launcher
// =====================================================================
extern "C" void kernel_launch(void* const* d_in, const int* in_sizes, int n_in,
                              void* d_out, int out_size, void* d_ws, size_t ws_size,
                              hipStream_t stream)
{
    (void)in_sizes; (void)n_in; (void)out_size; (void)ws_size;
    const float* xyz = (const float*)d_in[0];
    const float* nrm = (const float*)d_in[1];
    const float* w1 = (const float*)d_in[2];  const float* b1 = (const float*)d_in[3];
    const float* g1 = (const float*)d_in[4];  const float* e1 = (const float*)d_in[5];
    const float* w2 = (const float*)d_in[6];  const float* b2 = (const float*)d_in[7];
    const float* g2 = (const float*)d_in[8];  const float* e2 = (const float*)d_in[9];
    const float* w3 = (const float*)d_in[10]; const float* b3 = (const float*)d_in[11];
    const float* g3 = (const float*)d_in[12]; const float* e3 = (const float*)d_in[13];
    const float* w4 = (const float*)d_in[14]; const float* b4 = (const float*)d_in[15];
    const float* g4 = (const float*)d_in[16]; const float* e4 = (const float*)d_in[17];
    const float* w5 = (const float*)d_in[18]; const float* b5 = (const float*)d_in[19];
    const float* g5 = (const float*)d_in[20]; const float* e5 = (const float*)d_in[21];
    const float* w6 = (const float*)d_in[22]; const float* b6 = (const float*)d_in[23];
    float* out = (float*)d_out;
    char* ws = (char*)d_ws;

    const size_t S1 = (size_t)BB * KNN * NPTS;   // 262144 columns (layers 1-3)
    const size_t S2 = (size_t)BB * NPTS;         // 4096 columns (layers 4-6)

    // Region A (aliased): idx + F + buf1 live early; buf3 (256MB) reuses whole region.
    const size_t szA     = S1 * 512 * 2;                 // 268,435,456
    const size_t off_idx = 0;
    const size_t off_F   = off_idx + (size_t)BB*NPTS*KNN*4;  // 1 MB
    const size_t off_b1  = off_F + S1 * 32 * 2;              // +16 MB
    const size_t off_b3  = 0;
    size_t cur = szA;
    auto bump = [&](size_t bytes) { size_t o = cur; cur += (bytes + 255) & ~(size_t)255; return o; };
    const size_t off_b2 = bump(S1 * 256 * 2);
    const size_t off_b4 = bump(S2 * 512 * 2);
    const size_t off_b5 = bump(S2 * 256 * 2);
    const size_t off_b6 = bump(S2 * 256 * 2);
    const size_t off_w1 = bump(256 * 32  * 2);
    const size_t off_w2 = bump(256 * 256 * 2);
    const size_t off_w3 = bump(512 * 256 * 2);
    const size_t off_w4 = bump(256 * 512 * 2);
    const size_t off_w5 = bump(256 * 256 * 2);
    const size_t off_w6 = bump(256 * 256 * 2);
    const size_t off_st = bump(5 * BB * 8 * 2 * sizeof(double));   // peak ~412 MB

    int* idxp = (int*)(ws + off_idx);
    __hip_bfloat16* Fp = (__hip_bfloat16*)(ws + off_F);
    __hip_bfloat16* A1 = (__hip_bfloat16*)(ws + off_b1);
    __hip_bfloat16* A2 = (__hip_bfloat16*)(ws + off_b2);
    __hip_bfloat16* A3 = (__hip_bfloat16*)(ws + off_b3);
    __hip_bfloat16* A4 = (__hip_bfloat16*)(ws + off_b4);
    __hip_bfloat16* A5 = (__hip_bfloat16*)(ws + off_b5);
    __hip_bfloat16* A6 = (__hip_bfloat16*)(ws + off_b6);
    __hip_bfloat16* W1 = (__hip_bfloat16*)(ws + off_w1);
    __hip_bfloat16* W2 = (__hip_bfloat16*)(ws + off_w2);
    __hip_bfloat16* W3 = (__hip_bfloat16*)(ws + off_w3);
    __hip_bfloat16* W4 = (__hip_bfloat16*)(ws + off_w4);
    __hip_bfloat16* W5 = (__hip_bfloat16*)(ws + off_w5);
    __hip_bfloat16* W6 = (__hip_bfloat16*)(ws + off_w6);
    double* st  = (double*)(ws + off_st);
    double* st1 = st + 0 * BB * 16;
    double* st2 = st + 1 * BB * 16;
    double* st3 = st + 2 * BB * 16;
    double* st4 = st + 3 * BB * 16;
    double* st5 = st + 4 * BB * 16;

    hipMemsetAsync(ws + off_st, 0, 5 * BB * 8 * 2 * sizeof(double), stream);

    // weight conversion (cheap, every launch: deterministic, no cached state)
    cvt_w_k<<<(256*32  + 255)/256, 256, 0, stream>>>(w1, W1, 256, 11,  32);
    cvt_w_k<<<(256*256 + 255)/256, 256, 0, stream>>>(w2, W2, 256, 256, 256);
    cvt_w_k<<<(512*256 + 255)/256, 256, 0, stream>>>(w3, W3, 512, 256, 256);
    cvt_w_k<<<(256*512 + 255)/256, 256, 0, stream>>>(w4, W4, 256, 512, 512);
    cvt_w_k<<<(256*256 + 255)/256, 256, 0, stream>>>(w5, W5, 256, 256, 256);
    cvt_w_k<<<(256*256 + 255)/256, 256, 0, stream>>>(w6, W6, 256, 256, 256);

    ball_query_k<<<(BB*NPTS)/8, 256, 0, stream>>>(xyz, idxp);
    fused_features_k<<<(unsigned)((S1 + 255)/256), 256, 0, stream>>>(xyz, nrm, idxp, Fp);

    const float ic1 = 1.f / (32.f * 65536.f);
    const float ic3 = 1.f / (64.f * 65536.f);
    const float ic4 = 1.f / (32.f * 1024.f);

    // layer 1: 11(pad32) -> 256
    gemm_wmma_k<false><<<dim3((unsigned)(S1/256), 2), 256, 0, stream>>>(Fp, W1, b1, A1, 32, 256);
    gn_stats_k<<<dim3(64, 8, BB), 256, 0, stream>>>(A1, st1, 256, 32, 1024, 65536);
    gn_norm_relu_k<<<(unsigned)((S1*32 + 255)/256), 256, 0, stream>>>(A1, st1, g1, e1, 256, 32, 65536, S1, ic1);

    // layer 2: 256 -> 256
    gemm_wmma_k<false><<<dim3((unsigned)(S1/256), 2), 256, 0, stream>>>(A1, W2, b2, A2, 256, 256);
    gn_stats_k<<<dim3(64, 8, BB), 256, 0, stream>>>(A2, st2, 256, 32, 1024, 65536);
    gn_norm_relu_k<<<(unsigned)((S1*32 + 255)/256), 256, 0, stream>>>(A2, st2, g2, e2, 256, 32, 65536, S1, ic1);

    // layer 3: 256 -> 512 (buf3 aliases region A; inputs there are dead)
    gemm_wmma_k<false><<<dim3((unsigned)(S1/256), 4), 256, 0, stream>>>(A2, W3, b3, A3, 256, 512);
    gn_stats_k<<<dim3(64, 8, BB), 256, 0, stream>>>(A3, st3, 512, 64, 1024, 65536);
    gn3_max_k<<<(BB*NPTS*64)/256, 256, 0, stream>>>(A3, st3, g3, e3, A4, ic3);

    // layer 4: 512 -> 256
    gemm_wmma_k<false><<<dim3((unsigned)(S2/256), 2), 256, 0, stream>>>(A4, W4, b4, A5, 512, 256);
    gn_stats_k<<<dim3(1, 8, BB), 256, 0, stream>>>(A5, st4, 256, 32, 1024, 1024);
    gn_norm_relu_k<<<(unsigned)((S2*32 + 255)/256), 256, 0, stream>>>(A5, st4, g4, e4, 256, 32, 1024, S2, ic4);

    // layer 5: 256 -> 256
    gemm_wmma_k<false><<<dim3((unsigned)(S2/256), 2), 256, 0, stream>>>(A5, W5, b5, A6, 256, 256);
    gn_stats_k<<<dim3(1, 8, BB), 256, 0, stream>>>(A6, st5, 256, 32, 1024, 1024);
    gn_norm_relu_k<<<(unsigned)((S2*32 + 255)/256), 256, 0, stream>>>(A6, st5, g5, e5, 256, 32, 1024, S2, ic4);

    // layer 6: 256 -> 256, f32 straight into d_out ([B,N,256] == [col][cout])
    gemm_wmma_k<true><<<dim3((unsigned)(S2/256), 2), 256, 0, stream>>>(A6, W6, b6, out, 256, 256);
}
